// SeparateWindowedValuesCoordinatesAttention_43800076485206
// MI455X (gfx1250) — compile-verified, matrix operand order checked by reading, not run on hardware
//
#include <hip/hip_runtime.h>
#include <hip/hip_bf16.h>

typedef __attribute__((ext_vector_type(16))) __bf16 v16bf;
typedef __attribute__((ext_vector_type(8)))  __bf16 v8bf;
typedef __attribute__((ext_vector_type(8)))  float  v8f;

__device__ inline v8f vzero8() {
    v8f z;
#pragma unroll
    for (int i = 0; i < 8; ++i) z[i] = 0.0f;
    return z;
}

__device__ inline v8f wmma_bf16(v16bf a, v16bf b, v8f c) {
    return __builtin_amdgcn_wmma_f32_16x16x32_bf16(false, a, false, b, (short)0, c, false, false);
}

// Load a 16x32 bf16 WMMA A-fragment (also used for B via B^T layout) from LDS.
// sm is row-major [row][k] with stride `stride` (in bf16 elements, multiple of 8).
// Lane L: row = row_base + (L&15); elements 0..7 = K kh..kh+7, 8..15 = K kh+16..kh+23,
// where kh = k_base + 8*(L>>4).   (CDNA5 ISA 7.12.2, 16-bit A-matrix layout)
__device__ inline v16bf load_frag(const __bf16* sm, int stride, int row_base, int k_base) {
    const int lane = threadIdx.x & 31;
    const int r  = row_base + (lane & 15);
    const int kh = k_base + ((lane >> 4) << 3);
    const __bf16* p = sm + r * stride + kh;
    v8bf lo = *(const v8bf*)p;
    v8bf hi = *(const v8bf*)(p + 16);
    v16bf a;
#pragma unroll
    for (int i = 0; i < 8; ++i) { a[i] = lo[i]; a[i + 8] = hi[i]; }
    return a;
}

// ---------------------------------------------------------------------------
// Generic tiled WMMA GEMM:  C[M,N] = A[M,K] @ W[K,N] + bias
// A: fp32 (converted to bf16 on load) or bf16.  W,bias: fp32.  C: bf16 or fp32.
// Block = 256 threads (8 waves).  Tile: BM=64, BN=128, BK=32.
// Wave w computes the 16x64 strip (m0=(w&3)*16, n0=(w>>2)*64) = 4 C tiles.
// ---------------------------------------------------------------------------
template <bool IN_BF16, bool OUT_F32>
__global__ __launch_bounds__(256) void wmma_gemm(const void* __restrict__ Ap,
                                                 const float* __restrict__ W,
                                                 const float* __restrict__ bias,
                                                 void* __restrict__ Cp,
                                                 int M, int K, int N) {
    __shared__ __attribute__((aligned(16))) __bf16 As[64 * 40];   // stride 40 (80B rows)
    __shared__ __attribute__((aligned(16))) __bf16 Bt[128 * 40];  // B^T tile [n][k]

    const int tid  = threadIdx.x;
    const int wave = tid >> 5;
    const int lane = tid & 31;
    const int m0 = (wave & 3) << 4;
    const int n0 = (wave >> 2) << 6;
    const int mblk = blockIdx.y;
    const int nblk = blockIdx.x;

    v8f acc[4];
#pragma unroll
    for (int t = 0; t < 4; ++t) acc[t] = vzero8();

    const int arow = tid >> 2;
    const int akc  = (tid & 3) << 3;
    const int bk   = tid >> 3;
    const int bnc  = (tid & 7) << 4;

    for (int k0 = 0; k0 < K; k0 += 32) {
        // ---- stage A tile (64x32) into LDS as bf16 ----
        {
            v8bf pack;
            if (IN_BF16) {
                const __bf16* A = (const __bf16*)Ap;
                pack = *(const v8bf*)(A + (size_t)(mblk * 64 + arow) * K + k0 + akc);
            } else {
                const float* A = (const float*)Ap;
                const float* src = A + (size_t)(mblk * 64 + arow) * K + k0 + akc;
                float4 f0 = *(const float4*)src;
                float4 f1 = *(const float4*)(src + 4);
                pack[0] = (__bf16)f0.x; pack[1] = (__bf16)f0.y;
                pack[2] = (__bf16)f0.z; pack[3] = (__bf16)f0.w;
                pack[4] = (__bf16)f1.x; pack[5] = (__bf16)f1.y;
                pack[6] = (__bf16)f1.z; pack[7] = (__bf16)f1.w;
            }
            *(v8bf*)(&As[arow * 40 + akc]) = pack;
        }
        // ---- stage W tile (32x128) transposed into LDS ----
        {
            const float* src = W + (size_t)(k0 + bk) * N + nblk * 128 + bnc;
            if (k0 + 32 < K)
                __builtin_prefetch(src + (size_t)32 * N, 0, 1);  // global_prefetch_b8
#pragma unroll
            for (int c = 0; c < 4; ++c) {
                float4 f = *(const float4*)(src + c * 4);
                Bt[(bnc + c * 4 + 0) * 40 + bk] = (__bf16)f.x;
                Bt[(bnc + c * 4 + 1) * 40 + bk] = (__bf16)f.y;
                Bt[(bnc + c * 4 + 2) * 40 + bk] = (__bf16)f.z;
                Bt[(bnc + c * 4 + 3) * 40 + bk] = (__bf16)f.w;
            }
        }
        __syncthreads();
        v16bf a = load_frag(As, 40, m0, 0);
#pragma unroll
        for (int t = 0; t < 4; ++t) {
            v16bf b = load_frag(Bt, 40, n0 + t * 16, 0);
            acc[t] = wmma_bf16(a, b, acc[t]);
        }
        __syncthreads();
    }

    // ---- epilogue: C tile layout lane->N=lane&15, vgpr i -> M=i+8*(lane>>4) ----
    const int ncol_lo = lane & 15;
    const int rbase   = (lane >> 4) << 3;
#pragma unroll
    for (int t = 0; t < 4; ++t) {
        const int col = nblk * 128 + n0 + t * 16 + ncol_lo;
        const float bv = bias[col];
#pragma unroll
        for (int i = 0; i < 8; ++i) {
            const int row = mblk * 64 + m0 + rbase + i;
            const float v = acc[t][i] + bv;
            if (OUT_F32) ((float*)Cp)[(size_t)row * N + col] = v;
            else         ((__bf16*)Cp)[(size_t)row * N + col] = (__bf16)v;
        }
    }
}

// ---------------------------------------------------------------------------
// Per-token RMSNorm + interleaved q/k/v gather + roll/window/head relayout.
// One block per token (256 threads).
// Output layout per tensor: [b][head][window][token64][j48] bf16.
// ---------------------------------------------------------------------------
__global__ __launch_bounds__(256) void norm_gather(
    const __bf16* __restrict__ vqk_raw, const __bf16* __restrict__ vv_raw,
    const __bf16* __restrict__ cqk_raw,
    const float* __restrict__ g_vqk, const float* __restrict__ g_cqk,
    __bf16* __restrict__ qv, __bf16* __restrict__ kv, __bf16* __restrict__ vv,
    __bf16* __restrict__ qc, __bf16* __restrict__ kc) {
    __shared__ float sv[768];
    __shared__ float sc[768];
    __shared__ float svv[384];
    __shared__ float red[512];

    const int p   = blockIdx.x;
    const int tid = threadIdx.x;
    const int b   = p / (192 * 192);
    const int rem = p - b * 192 * 192;
    const int h   = rem / 192;
    const int w   = rem - h * 192;

    float ss_v = 0.f, ss_c = 0.f;
    for (int i = tid; i < 768; i += 256) {
        float x = (float)vqk_raw[(size_t)p * 768 + i];
        sv[i] = x; ss_v += x * x;
        float y = (float)cqk_raw[(size_t)p * 768 + i];
        sc[i] = y; ss_c += y * y;
    }
    for (int i = tid; i < 384; i += 256)
        svv[i] = (float)vv_raw[(size_t)p * 384 + i];

    red[tid] = ss_v;
    red[256 + tid] = ss_c;
    __syncthreads();
    for (int s = 128; s > 0; s >>= 1) {
        if (tid < s) { red[tid] += red[tid + s]; red[256 + tid] += red[256 + tid + s]; }
        __syncthreads();
    }
    const float r1 = rsqrtf(red[0] / 768.f + 1e-6f);
    const float r2 = rsqrtf(red[256] / 768.f + 1e-6f);

    // roll by -4: original (h,w) lands at rolled position ((h-4)%192,(w-4)%192)
    const int hp  = (h + 188) % 192;
    const int wp  = (w + 188) % 192;
    const int win = (hp >> 3) * 24 + (wp >> 3);
    const int t   = ((hp & 7) << 3) + (wp & 7);

    for (int i = tid; i < 1920; i += 256) {
        const int tensor = i / 384;
        const int r    = i - tensor * 384;
        const int head = r / 48;
        const int j    = r - head * 48;
        const int d    = j * 8 + head;  // feature split (e, NH), NH fast
        const size_t dst = ((((size_t)b * 8 + head) * 576 + win) * 64 + t) * 48 + j;
        float val;
        if (tensor < 3) {  // interleaved (ID,3): index 3d+k over cat(vqk_norm, vv)
            const int c = 3 * d + tensor;
            val = (c < 768) ? sv[c] * r1 * g_vqk[c] : svv[c - 768];
        } else {           // coords interleaved (ID,2)
            const int c = 2 * d + (tensor - 3);
            val = sc[c] * r2 * g_cqk[c];
        }
        const __bf16 bvv = (__bf16)val;
        switch (tensor) {
            case 0: qv[dst] = bvv; break;
            case 1: kv[dst] = bvv; break;
            case 2: vv[dst] = bvv; break;
            case 3: qc[dst] = bvv; break;
            default: kc[dst] = bvv; break;
        }
    }
}

// ---------------------------------------------------------------------------
// Windowed attention: one block per (b,head,window).  256 threads = 8 waves.
// S = (Q K^T + lambda Qc Kc^T)/sqrt(48) + rel_bias + shift masks; softmax; P V.
// ---------------------------------------------------------------------------
__global__ __launch_bounds__(256) void win_attn(
    const __bf16* __restrict__ qv, const __bf16* __restrict__ kv,
    const __bf16* __restrict__ vvp, const __bf16* __restrict__ qc,
    const __bf16* __restrict__ kc, const float* __restrict__ pos_emb,
    const float* __restrict__ att_lambda, __bf16* __restrict__ y) {
    __shared__ __attribute__((aligned(16))) __bf16 Qs[64 * 72];
    __shared__ __attribute__((aligned(16))) __bf16 Ks[64 * 72];
    __shared__ __attribute__((aligned(16))) __bf16 Qcs[64 * 72];
    __shared__ __attribute__((aligned(16))) __bf16 Kcs[64 * 72];
    __shared__ __attribute__((aligned(16))) __bf16 Vt[48 * 72];  // transposed V: [e][token]
    __shared__ __attribute__((aligned(16))) __bf16 P[64 * 72];   // softmax probs bf16
    __shared__ float S[64 * 68];
    __shared__ float pe[225];

    const int blk  = blockIdx.x;        // (b*8+head)*576 + win
    const int win  = blk % 576;
    const int bh   = blk / 576;
    const int head = bh & 7;
    const int bidx = bh >> 3;
    const int wh   = win / 24;
    const int ww   = win - wh * 24;
    const int tid  = threadIdx.x;
    const int wave = tid >> 5;
    const int lane = tid & 31;
    const size_t gbase = (size_t)blk * (64 * 48);

    // ---- stage tiles: 64 rows x 48 bf16 = 6 b128 chunks per row ----
    for (int idx = tid; idx < 64 * 6; idx += 256) {
        const int r = idx / 6, c = idx - r * 6;
        const size_t g = gbase + (size_t)r * 48 + c * 8;
        const int  l = r * 72 + c * 8;
        *(v8bf*)(&Qs[l])  = *(const v8bf*)(qv + g);
        *(v8bf*)(&Ks[l])  = *(const v8bf*)(kv + g);
        *(v8bf*)(&Qcs[l]) = *(const v8bf*)(qc + g);
        *(v8bf*)(&Kcs[l]) = *(const v8bf*)(kc + g);
        v8bf v = *(const v8bf*)(vvp + g);
#pragma unroll
        for (int i = 0; i < 8; ++i) Vt[(c * 8 + i) * 72 + r] = v[i];  // transpose store
    }
    // zero-pad head-dim 48..63 for the K=64 WMMA loop
    {
        v8bf z;
#pragma unroll
        for (int i = 0; i < 8; ++i) z[i] = (__bf16)0.0f;
        for (int idx = tid; idx < 64 * 2; idx += 256) {
            const int r = idx >> 1, c = 6 + (idx & 1);
            const int l = r * 72 + c * 8;
            *(v8bf*)(&Qs[l]) = z; *(v8bf*)(&Ks[l]) = z;
            *(v8bf*)(&Qcs[l]) = z; *(v8bf*)(&Kcs[l]) = z;
        }
    }
    if (tid < 225) pe[tid] = pos_emb[tid];
    const float lam = att_lambda[0];
    __syncthreads();

    // ---- scores: 16 tiles of 16x16, 2 per wave ----
    const float scale = 0.14433756729740643f;  // 1/sqrt(48)
    const bool mrow = (wh == 23);
    const bool mcol = (ww == 23);
#pragma unroll
    for (int tt = 0; tt < 2; ++tt) {
        const int tile = wave * 2 + tt;
        const int mi = (tile >> 2) << 4;
        const int nj = (tile & 3) << 4;
        v8f accv = vzero8(), accc = vzero8();
#pragma unroll
        for (int ks = 0; ks < 64; ks += 32) {
            v16bf aq = load_frag(Qs, 72, mi, ks);
            v16bf bk = load_frag(Ks, 72, nj, ks);   // B frag = A-layout of K rows
            accv = wmma_bf16(aq, bk, accv);
            v16bf ac = load_frag(Qcs, 72, mi, ks);
            v16bf bc = load_frag(Kcs, 72, nj, ks);
            accc = wmma_bf16(ac, bc, accc);
        }
        const int col = nj + (lane & 15);
        const int xj = col >> 3, yj = col & 7;
#pragma unroll
        for (int i = 0; i < 8; ++i) {
            const int row = mi + ((lane >> 4) << 3) + i;
            const int xi = row >> 3, yi = row & 7;
            float s = (accv[i] + lam * accc[i]) * scale;
            s += pe[(xj - xi + 7) * 15 + (yj - yi + 7)];
            if (mrow && ((xi >= 4) != (xj >= 4))) s = -1e30f;
            if (mcol && ((yi >= 4) != (yj >= 4))) s = -1e30f;
            S[row * 68 + col] = s;
        }
    }
    __syncthreads();

    // ---- softmax (one row per thread, 64 rows) ----
    if (tid < 64) {
        float mx = -1e38f;
        for (int j = 0; j < 64; ++j) mx = fmaxf(mx, S[tid * 68 + j]);
        float sum = 0.f;
        for (int j = 0; j < 64; ++j) {
            float e = __expf(S[tid * 68 + j] - mx);
            sum += e;
            S[tid * 68 + j] = e;
        }
        const float inv = 1.f / sum;
        for (int j = 0; j < 64; ++j) P[tid * 72 + j] = (__bf16)(S[tid * 68 + j] * inv);
    }
    __syncthreads();

    // ---- O = P @ V : 4x3 tiles of 16x16 over 8 waves ----
    for (int tile = wave; tile < 12; tile += 8) {
        const int mi = (tile / 3) << 4;
        const int nj = (tile % 3) << 4;
        v8f acc = vzero8();
#pragma unroll
        for (int ks = 0; ks < 64; ks += 32) {
            v16bf a = load_frag(P, 72, mi, ks);
            v16bf b = load_frag(Vt, 72, nj, ks);    // B frag from transposed V
            acc = wmma_bf16(a, b, acc);
        }
        const int e = nj + (lane & 15);
        const int feat = head * 48 + e;             // output layout (H e): head slow
#pragma unroll
        for (int i = 0; i < 8; ++i) {
            const int t  = mi + ((lane >> 4) << 3) + i;
            const int hp = wh * 8 + (t >> 3);
            const int wp = ww * 8 + (t & 7);
            const size_t yi = (((size_t)bidx * 192 + hp) * 192 + wp) * 384 + feat;
            y[yi] = (__bf16)acc[i];
        }
    }
}

extern "C" void kernel_launch(void* const* d_in, const int* in_sizes, int n_in,
                              void* d_out, int out_size, void* d_ws, size_t ws_size,
                              hipStream_t stream) {
    (void)in_sizes; (void)n_in; (void)out_size; (void)ws_size;
    const float* values     = (const float*)d_in[0];
    const float* coords     = (const float*)d_in[1];
    const float* W_vqk      = (const float*)d_in[2];
    const float* b_vqk      = (const float*)d_in[3];
    const float* g_vqk      = (const float*)d_in[4];
    const float* W_vv       = (const float*)d_in[5];
    const float* b_vv       = (const float*)d_in[6];
    const float* W_cqk      = (const float*)d_in[7];
    const float* b_cqk      = (const float*)d_in[8];
    const float* g_cqk      = (const float*)d_in[9];
    const float* att_lambda = (const float*)d_in[10];
    const float* pos_emb    = (const float*)d_in[11];
    const float* W_ov       = (const float*)d_in[12];
    const float* b_ov       = (const float*)d_in[13];

    const size_t T = 2ull * 192 * 192;          // 73728 tokens
    const size_t QSZ = 2ull * 8 * 576 * 64 * 48;  // elements per windowed tensor

    char* ws = (char*)d_ws;
    __bf16* vqk_raw = (__bf16*)ws;                                   // T*768
    __bf16* vv_raw  = (__bf16*)(ws + T * 768 * 2);                   // T*384
    __bf16* cqk_raw = (__bf16*)(ws + T * 768 * 2 + T * 384 * 2);     // T*768
    __bf16* qv = (__bf16*)(ws + T * 768 * 2 * 2 + T * 384 * 2);
    __bf16* kv = qv + QSZ;
    __bf16* vvb = kv + QSZ;
    __bf16* qc = vvb + QSZ;
    __bf16* kc = qc + QSZ;
    __bf16* y  = (__bf16*)ws;  // alias vqk_raw region (dead after norm_gather)

    // 1) projections (bf16 out)
    wmma_gemm<false, false><<<dim3(6, T / 64), 256, 0, stream>>>(
        values, W_vqk, b_vqk, vqk_raw, (int)T, 384, 768);
    wmma_gemm<false, false><<<dim3(3, T / 64), 256, 0, stream>>>(
        values, W_vv, b_vv, vv_raw, (int)T, 384, 384);
    wmma_gemm<false, false><<<dim3(6, T / 64), 256, 0, stream>>>(
        coords, W_cqk, b_cqk, cqk_raw, (int)T, 128, 768);

    // 2) RMSNorm + interleave gather + roll/window/head relayout
    norm_gather<<<dim3((unsigned)T), 256, 0, stream>>>(
        vqk_raw, vv_raw, cqk_raw, g_vqk, g_cqk, qv, kv, vvb, qc, kc);

    // 3) attention per (b, head, window)
    win_attn<<<dim3(2 * 8 * 576), 256, 0, stream>>>(
        qv, kv, vvb, qc, kc, pos_emb, att_lambda, y);

    // 4) output projection (fp32 out)
    wmma_gemm<true, true><<<dim3(3, T / 64), 256, 0, stream>>>(
        y, W_ov, b_ov, d_out, (int)T, 384, 384);
}
